// SummarizerModel_2903397892533
// MI455X (gfx1250) — compile-verified, matrix operand order checked by reading, not run on hardware
//
#include <hip/hip_runtime.h>
#include <hip/hip_bf16.h>
#include <math.h>

// ---------------------------------------------------------------------------
// Model constants
// ---------------------------------------------------------------------------
#define TT   2049          // tokens
#define TP   2080          // padded tokens (multiple of 32)
#define DM   1024          // model dim
#define HH   16            // heads
#define DHD  64            // head dim
#define D3   3072          // 3*D
#define LL   6             // layers
#define DFF_ 4096
#define SS   64            // sentences
#define SENT_ 31
#define PSTR 2096          // LDS row stride (floats) to avoid bank conflicts

typedef __attribute__((ext_vector_type(16))) __bf16 v16bf;
typedef __attribute__((ext_vector_type(8)))  float  v8f;

// single dynamic-LDS symbol shared by all kernels
extern __shared__ char dyn_lds[];

// ---------------------------------------------------------------------------
// 16-bit A/B fragment loader (wave32).  ISA layout for 16-bit 16x32 A-matrix:
//   lanes 0-15 : row = lane,    K = {kb+0..7} then {kb+16..23}
//   lanes 16-31: row = lane-16, K = {kb+8..15} then {kb+24..31}
// Row base must be 16B aligned (true everywhere we use it).
// ---------------------------------------------------------------------------
__device__ __forceinline__ v16bf load_frag(const __bf16* base, int ld, int kb) {
    const int l  = threadIdx.x & 31;
    const int r  = l & 15;
    const int kh = (l >> 4) << 3;
    const __bf16* p = base + (size_t)r * ld + kb + kh;
    union { uint4 u[2]; v16bf v; } f;
    f.u[0] = *(const uint4*)(p);        // K: kb+kh .. kb+kh+7
    f.u[1] = *(const uint4*)(p + 16);   // K: kb+16+kh .. kb+16+kh+7
    return f.v;
}

// ---------------------------------------------------------------------------
// Async global -> LDS 16-byte copy (CDNA5), tracked by ASYNCcnt.
// ldsOff = byte offset of destination inside the workgroup LDS allocation.
// ---------------------------------------------------------------------------
__device__ __forceinline__ void async_copy16(unsigned ldsOff, const void* gptr) {
    asm volatile("global_load_async_to_lds_b128 %0, %1, off"
                 :: "v"(ldsOff), "v"(gptr)
                 : "memory");
}
__device__ __forceinline__ void wait_async0() {
    asm volatile("s_wait_asynccnt 0x0" ::: "memory");
}

// ---------------------------------------------------------------------------
// Generic WMMA GEMM:  out[M,N] = A[M,K] (bf16) @ W[N,K]^T (bf16) + bias
//                     (+ resid), optional ReLU, f32 and/or bf16 stores.
// grid = (M/16, N/128), block = 256 (8 waves, one 16x16 tile per wave).
// The 16xK A tile is staged into LDS once per block with async-to-LDS copies
// (removes the 8x redundant A reads the naive version does), then each wave
// runs the WMMA K-loop with A fragments from LDS and B streamed from L2.
// Dynamic LDS = 16*K*2 bytes (32 KB for K=1024, 128 KB for K=4096).
// ---------------------------------------------------------------------------
__global__ __launch_bounds__(256)
void gemm_wmma_kernel(const __bf16* __restrict__ A, const __bf16* __restrict__ W,
                      const float* __restrict__ bias, const float* __restrict__ resid,
                      float* __restrict__ outF, __bf16* __restrict__ outB,
                      int M, int N, int K, int relu)
{
    __bf16* aTile = (__bf16*)dyn_lds;                 // [16][K]
    const int wave = threadIdx.x >> 5;
    const int mt   = blockIdx.x;
    const int nt   = blockIdx.y * 8 + wave;
    const bool valid = (nt * 16 < N);

    // ---- stage A tile: global -> LDS via async copies --------------------
    {
        const char* Ab = (const char*)(A + (size_t)mt * 16 * K);
        const unsigned ldsBase = (unsigned)(size_t)(void*)aTile;
        const unsigned abytes  = (unsigned)16 * (unsigned)K * 2u;
        for (unsigned byte = (unsigned)threadIdx.x * 16u; byte < abytes; byte += 256u * 16u)
            async_copy16(ldsBase + byte, Ab + byte);
        wait_async0();
    }
    __syncthreads();

    if (valid) {
        const __bf16* Wb = W + (size_t)nt * 16 * K;
        v8f acc = {};
        for (int k = 0; k < K; k += 32) {
            v16bf a = load_frag(aTile, K, k);         // LDS
            v16bf b = load_frag(Wb, K, k);            // L2/global
            acc = __builtin_amdgcn_wmma_f32_16x16x32_bf16(
                      false, a, false, b, (short)0, acc, false, false);
        }

        const int l     = threadIdx.x & 31;
        const int n     = nt * 16 + (l & 15);
        const int mbase = mt * 16 + ((l >> 4) << 3);
        const float bv  = bias ? bias[n] : 0.0f;
#pragma unroll
        for (int r = 0; r < 8; ++r) {
            const size_t idx = (size_t)(mbase + r) * N + n;
            float v = acc[r] + bv;
            if (resid) v += resid[idx];
            if (relu)  v = fmaxf(v, 0.0f);
            if (outF)  outF[idx] = v;
            if (outB)  outB[idx] = (__bf16)v;
        }
    }
}

// ---------------------------------------------------------------------------
// Embedding + sinusoidal positional encoding.  grid = TP, block = 256.
// Padded rows (t >= TT) are zeroed.
// ---------------------------------------------------------------------------
__global__ __launch_bounds__(256)
void embed_kernel(const int* __restrict__ ids, const float* __restrict__ emb,
                  float* __restrict__ xf, __bf16* __restrict__ xb)
{
    const int t = blockIdx.x;
    for (int d = threadIdx.x; d < DM; d += 256) {
        float val = 0.0f;
        if (t < TT) {
            const float inv = __expf(-(float)(d & ~1) * (9.21034037198e0f / (float)DM));
            const float arg = (float)t * inv;
            const float pe  = (d & 1) ? __cosf(arg) : __sinf(arg);
            val = emb[(size_t)ids[t] * DM + d] * 32.0f + pe;   // sqrt(1024)=32
        }
        xf[(size_t)t * DM + d] = val;
        xb[(size_t)t * DM + d] = (__bf16)val;
    }
}

// ---------------------------------------------------------------------------
// f32 -> bf16 weight conversion
// ---------------------------------------------------------------------------
__global__ __launch_bounds__(256)
void cvt_bf16_kernel(const float* __restrict__ src, __bf16* __restrict__ dst, int n)
{
    int i = blockIdx.x * 256 + threadIdx.x;
    if (i < n) dst[i] = (__bf16)src[i];
}

// ---------------------------------------------------------------------------
// Build V^T:  vT[(h*64+dh)*TP + t] = qkv[t, 2D + h*64+dh]
// ---------------------------------------------------------------------------
__global__ __launch_bounds__(256)
void vtrans_kernel(const __bf16* __restrict__ qkvb, __bf16* __restrict__ vT)
{
    int i = blockIdx.x * 256 + threadIdx.x;
    if (i >= DM * TP) return;
    int c = i / TP;        // h*64+dh
    int t = i - c * TP;
    vT[i] = qkvb[(size_t)t * D3 + 2 * DM + c];
}

// ---------------------------------------------------------------------------
// Attention for one 16-query tile, all heads, one layer.
// block = 256 (8 waves); dynamic LDS = 16*PSTR floats (score/prob tile).
//  phase1: scores = Q Kt (WMMA, bf16), -inf mask for padded keys
//  phase2: softmax per row (wave32 shfl reductions) + head-mean -> attn map
//  phase3: out = P V  (WMMA; P from LDS cvt to bf16, V^T contiguous)
// ---------------------------------------------------------------------------
__global__ __launch_bounds__(256)
void attn_kernel(const __bf16* __restrict__ qkvb, const __bf16* __restrict__ vT,
                 __bf16* __restrict__ attn_ob, float* __restrict__ amap_base)
{
    float* probs = (float*)dyn_lds;           // [16][PSTR]
    const int wave = threadIdx.x >> 5;
    const int l    = threadIdx.x & 31;
    const int qt   = blockIdx.x;
    const float scale = 0.125f;               // 1/sqrt(64)

    for (int h = 0; h < HH; ++h) {
        // ---- phase 1: scores -------------------------------------------------
        const __bf16* Qb = qkvb + (size_t)qt * 16 * D3 + h * DHD;
        for (int nt = wave; nt < TP / 16; nt += 8) {
            const __bf16* Kb = qkvb + (size_t)nt * 16 * D3 + DM + h * DHD;
            v8f acc = {};
            for (int k = 0; k < DHD; k += 32) {
                v16bf a = load_frag(Qb, D3, k);
                v16bf b = load_frag(Kb, D3, k);
                acc = __builtin_amdgcn_wmma_f32_16x16x32_bf16(
                          false, a, false, b, (short)0, acc, false, false);
            }
            const int n     = nt * 16 + (l & 15);
            const int mbase = (l >> 4) << 3;
#pragma unroll
            for (int r = 0; r < 8; ++r)
                probs[(mbase + r) * PSTR + n] = (n < TT) ? acc[r] * scale : -INFINITY;
        }
        __syncthreads();

        // ---- phase 2: softmax + head-averaged attention map ------------------
        {
            const int row = wave * 2 + (l >> 4);
            const int sub = l & 15;
            float* pr = probs + row * PSTR;
            float mx = -INFINITY;
            for (int j = sub; j < TP; j += 16) mx = fmaxf(mx, pr[j]);
            for (int o = 8; o; o >>= 1) mx = fmaxf(mx, __shfl_xor(mx, o, 32));
            float sum = 0.0f;
            for (int j = sub; j < TP; j += 16) {
                float e = __expf(pr[j] - mx);
                pr[j] = e;
                sum += e;
            }
            for (int o = 8; o; o >>= 1) sum += __shfl_xor(sum, o, 32);
            const float inv  = 1.0f / sum;
            const int   qrow = qt * 16 + row;
            float* amap = amap_base + (size_t)qrow * TT;
            for (int j = sub; j < TP; j += 16) {
                float p = pr[j] * inv;
                pr[j] = p;
                if (qrow < TT && j < TT) {
                    float m = p * (1.0f / (float)HH);
                    amap[j] = (h == 0) ? m : (amap[j] + m);
                }
            }
        }
        __syncthreads();

        // ---- phase 3: out = P @ V  (waves 0..3, one 16-col DH tile each) -----
        if (wave < 4) {
            const __bf16* Vb = vT + (size_t)(h * DHD + wave * 16) * TP;
            v8f acc = {};
            const int r0 = l & 15;
            const int kh = (l >> 4) << 3;
            for (int kt = 0; kt < TP; kt += 32) {
                v16bf a;
#pragma unroll
                for (int i = 0; i < 8; ++i) {
                    a[i]     = (__bf16)probs[r0 * PSTR + kt + kh + i];
                    a[8 + i] = (__bf16)probs[r0 * PSTR + kt + 16 + kh + i];
                }
                v16bf b = load_frag(Vb, TP, kt);
                acc = __builtin_amdgcn_wmma_f32_16x16x32_bf16(
                          false, a, false, b, (short)0, acc, false, false);
            }
            const int col   = h * DHD + wave * 16 + (l & 15);
            const int mbase = qt * 16 + ((l >> 4) << 3);
#pragma unroll
            for (int r = 0; r < 8; ++r)
                attn_ob[(size_t)(mbase + r) * DM + col] = (__bf16)acc[r];
        }
        __syncthreads();
    }
}

// ---------------------------------------------------------------------------
// LayerNorm over D=1024; grid = TP (one row per block); padded rows zeroed.
// Writes both f32 (for residuals) and bf16 (for GEMM A operands).
// ---------------------------------------------------------------------------
__global__ __launch_bounds__(256)
void ln_kernel(const float* __restrict__ y, const float* __restrict__ g,
               const float* __restrict__ b, float* __restrict__ xf,
               __bf16* __restrict__ xb)
{
    __shared__ float red[256];
    const int t   = blockIdx.x;
    const int tid = threadIdx.x;
    if (t >= TT) {
        for (int d = tid; d < DM; d += 256) {
            xf[(size_t)t * DM + d] = 0.0f;
            xb[(size_t)t * DM + d] = (__bf16)0.0f;
        }
        return;
    }
    const float* yr = y + (size_t)t * DM;
    float s = 0.0f;
    for (int d = tid; d < DM; d += 256) s += yr[d];
    red[tid] = s; __syncthreads();
    for (int o = 128; o; o >>= 1) { if (tid < o) red[tid] += red[tid + o]; __syncthreads(); }
    const float mean = red[0] * (1.0f / DM);
    __syncthreads();
    float v = 0.0f;
    for (int d = tid; d < DM; d += 256) { float c = yr[d] - mean; v += c * c; }
    red[tid] = v; __syncthreads();
    for (int o = 128; o; o >>= 1) { if (tid < o) red[tid] += red[tid + o]; __syncthreads(); }
    const float rstd = rsqrtf(red[0] * (1.0f / DM) + 1e-5f);
    for (int d = tid; d < DM; d += 256) {
        float out = (yr[d] - mean) * rstd * g[d] + b[d];
        xf[(size_t)t * DM + d] = out;
        xb[(size_t)t * DM + d] = (__bf16)out;
    }
}

// ---------------------------------------------------------------------------
// Sentence mean pooling via seg_ids.  grid = SS, block = 256.
// ---------------------------------------------------------------------------
__global__ __launch_bounds__(256)
void pool_kernel(const float* __restrict__ xf, const int* __restrict__ seg,
                 float* __restrict__ sums)
{
    const int s = blockIdx.x;
    for (int d = threadIdx.x; d < DM; d += 256) {
        float acc = 0.0f;
        for (int t = 0; t < TT; ++t)
            if (seg[t] == s) acc += xf[(size_t)t * DM + d];
        sums[(size_t)s * DM + d] = acc * (1.0f / (float)SENT_);
    }
}

__global__ __launch_bounds__(256)
void scorer1_kernel(const float* __restrict__ sums, const float* __restrict__ w1,
                    const float* __restrict__ b1, float* __restrict__ hbuf)
{
    const int s = blockIdx.x;
    for (int j = threadIdx.x; j < DM / 2; j += 256) {
        float acc = b1[j];
        const float* sr = sums + (size_t)s * DM;
        const float* wr = w1 + (size_t)j * DM;
        for (int d = 0; d < DM; ++d) acc += sr[d] * wr[d];
        hbuf[s * (DM / 2) + j] = fmaxf(acc, 0.0f);
    }
}

__global__ __launch_bounds__(64)
void scorer2_kernel(const float* __restrict__ hbuf, const float* __restrict__ w2,
                    const float* __restrict__ b2, float* __restrict__ out)
{
    const int s = threadIdx.x;
    if (s >= SS) return;
    float acc = b2[0];
    for (int j = 0; j < DM / 2; ++j) acc += hbuf[s * (DM / 2) + j] * w2[j];
    out[s] = acc;
}

// ---------------------------------------------------------------------------
// Host-side launch
// ---------------------------------------------------------------------------
extern "C" void kernel_launch(void* const* d_in, const int* in_sizes, int n_in,
                              void* d_out, int out_size, void* d_ws, size_t ws_size,
                              hipStream_t stream)
{
    (void)in_sizes; (void)n_in; (void)out_size; (void)ws_size;

    const int*   ids        = (const int*)  d_in[0];
    const int*   seg        = (const int*)  d_in[1];
    const float* token_emb  = (const float*)d_in[2];
    const float* in_proj_w  = (const float*)d_in[3];
    const float* in_proj_b  = (const float*)d_in[4];
    const float* out_proj_w = (const float*)d_in[5];
    const float* out_proj_b = (const float*)d_in[6];
    const float* lin1_w     = (const float*)d_in[7];
    const float* lin1_b     = (const float*)d_in[8];
    const float* lin2_w     = (const float*)d_in[9];
    const float* lin2_b     = (const float*)d_in[10];
    const float* ln1_g      = (const float*)d_in[11];
    const float* ln1_b      = (const float*)d_in[12];
    const float* ln2_g      = (const float*)d_in[13];
    const float* ln2_b      = (const float*)d_in[14];
    const float* sc_w1      = (const float*)d_in[15];
    const float* sc_b1      = (const float*)d_in[16];
    const float* sc_w2      = (const float*)d_in[17];
    const float* sc_b2      = (const float*)d_in[18];

    float* out_scores = (float*)d_out;           // [64]
    float* out_attn   = out_scores + SS;         // [L, T, T]

    // ---- workspace carving ----------------------------------------------
    char* ws = (char*)d_ws;
    size_t off = 0;
    auto take = [&](size_t bytes) { char* p = ws + off; off += (bytes + 255) & ~(size_t)255; return p; };
    float*  xf      = (float*) take((size_t)TP * DM * 4);
    __bf16* xb      = (__bf16*)take((size_t)TP * DM * 2);
    float*  yf      = (float*) take((size_t)TP * DM * 4);
    __bf16* qkvb    = (__bf16*)take((size_t)TP * D3 * 2);
    __bf16* vT      = (__bf16*)take((size_t)DM * TP * 2);
    __bf16* attn_ob = (__bf16*)take((size_t)TP * DM * 2);
    __bf16* ff1b    = (__bf16*)take((size_t)TP * DFF_ * 2);
    __bf16* wb      = (__bf16*)take((size_t)DFF_ * DM * 2);   // largest weight slice
    float*  sums    = (float*) take((size_t)SS * DM * 4);
    float*  hbuf    = (float*) take((size_t)SS * (DM / 2) * 4);

    const dim3 blk(256);
    const int  MT = TP / 16;                 // 130 row tiles
    const size_t attn_shmem = (size_t)16 * PSTR * sizeof(float);
    const size_t gemm_shmem_k1024 = (size_t)16 * DM * 2;    // 32 KB
    const size_t gemm_shmem_k4096 = (size_t)16 * DFF_ * 2;  // 128 KB

    // ---- embedding + positional encoding --------------------------------
    embed_kernel<<<dim3(TP), blk, 0, stream>>>(ids, token_emb, xf, xb);

    for (int l = 0; l < LL; ++l) {
        // QKV projection
        { int n = D3 * DM;
          cvt_bf16_kernel<<<dim3((n + 255) / 256), blk, 0, stream>>>(in_proj_w + (size_t)l * n, wb, n); }
        gemm_wmma_kernel<<<dim3(MT, D3 / 128), blk, gemm_shmem_k1024, stream>>>(
            xb, wb, in_proj_b + (size_t)l * D3, nullptr, nullptr, qkvb, TP, D3, DM, 0);

        // V transpose for contiguous B fragments in attn@V
        { int n = DM * TP;
          vtrans_kernel<<<dim3((n + 255) / 256), blk, 0, stream>>>(qkvb, vT); }

        // attention (scores, softmax, head-mean map, P@V)
        attn_kernel<<<dim3(MT), blk, attn_shmem, stream>>>(
            qkvb, vT, attn_ob, out_attn + (size_t)l * TT * TT);

        // output projection + residual
        { int n = DM * DM;
          cvt_bf16_kernel<<<dim3((n + 255) / 256), blk, 0, stream>>>(out_proj_w + (size_t)l * n, wb, n); }
        gemm_wmma_kernel<<<dim3(MT, DM / 128), blk, gemm_shmem_k1024, stream>>>(
            attn_ob, wb, out_proj_b + (size_t)l * DM, xf, yf, nullptr, TP, DM, DM, 0);
        ln_kernel<<<dim3(TP), blk, 0, stream>>>(yf, ln1_g + (size_t)l * DM, ln1_b + (size_t)l * DM, xf, xb);

        // FFN
        { int n = DFF_ * DM;
          cvt_bf16_kernel<<<dim3((n + 255) / 256), blk, 0, stream>>>(lin1_w + (size_t)l * n, wb, n); }
        gemm_wmma_kernel<<<dim3(MT, DFF_ / 128), blk, gemm_shmem_k1024, stream>>>(
            xb, wb, lin1_b + (size_t)l * DFF_, nullptr, nullptr, ff1b, TP, DFF_, DM, 1);
        { int n = DM * DFF_;
          cvt_bf16_kernel<<<dim3((n + 255) / 256), blk, 0, stream>>>(lin2_w + (size_t)l * n, wb, n); }
        gemm_wmma_kernel<<<dim3(MT, DM / 128), blk, gemm_shmem_k4096, stream>>>(
            ff1b, wb, lin2_b + (size_t)l * DM, xf, yf, nullptr, TP, DM, DFF_, 0);
        ln_kernel<<<dim3(TP), blk, 0, stream>>>(yf, ln2_g + (size_t)l * DM, ln2_b + (size_t)l * DM, xf, xb);
    }

    // ---- pooling + scorer ----------------------------------------------
    pool_kernel<<<dim3(SS), blk, 0, stream>>>(xf, seg, sums);
    scorer1_kernel<<<dim3(SS), blk, 0, stream>>>(sums, sc_w1, sc_b1, hbuf);
    scorer2_kernel<<<dim3(1), dim3(64), 0, stream>>>(hbuf, sc_w2, sc_b2, out_scores);
}
